// MPNN_4913442586654
// MI455X (gfx1250) — matrix operand order repaired and checked by reading, not compiled, not run on hardware
//
#include <hip/hip_runtime.h>
#include <hip/hip_bf16.h>
#include <math.h>

// ---------------------------------------------------------------------------
// Types for CDNA5 WMMA (wave32): 16x16x32 bf16 -> f32
// ---------------------------------------------------------------------------
typedef __attribute__((ext_vector_type(16))) __bf16 v16bf;
typedef __attribute__((ext_vector_type(8)))  __bf16 v8bf;
typedef __attribute__((ext_vector_type(8)))  float  v8f;

__device__ __forceinline__ __bf16 f2bf(float f) {
  unsigned u = __builtin_bit_cast(unsigned, f);
  unsigned short h = (unsigned short)(u >> 16);   // truncate to bf16
  return __builtin_bit_cast(__bf16, h);
}

// ---------------------------------------------------------------------------
// Y = act(X @ W^T + b): X:[R,K] bf16 row-major, W:[C,K] bf16 row-major.
// Optional f32 output Yf and/or bf16 output Yb. R % 16 == 0, K % 32 == 0,
// C % 16 == 0 are guaranteed by construction (inputs padded in workspace),
// so there are no bounds checks and all fragment loads are contiguous
// vector loads (A: 2 x 16B per lane, B: 1 x 32B per lane).
//
// A fragment (16x32, 16-bit): lanes 0-15 row M=lane%16, slots j -> K=j;
// lanes 16-31 same rows, K offset +16 (K=16+j).
// B fragment (32x16, 16-bit): lane col N=lane%16; lanes 0-15 K=j,
// lanes 16-31 K=16+j.
// C/D f32: VGPR i -> row i + 8*(lane/16), col = lane%16.
// ---------------------------------------------------------------------------
template<bool RELU, bool BIAS>
__global__ __launch_bounds__(32)
void gemm_bf16_wmma(const __bf16* __restrict__ X, const __bf16* __restrict__ W,
                    const float* __restrict__ bias,
                    float* __restrict__ Yf, __bf16* __restrict__ Yb,
                    int R, int K, int C)
{
  const int ct   = blockIdx.x;
  const int rt   = blockIdx.y;
  const int lane = threadIdx.x;
  const int l15  = lane & 15;
  const int grp  = lane >> 4;
  const int row  = rt * 16 + l15;   // A-fragment row
  const int col  = ct * 16 + l15;   // B-fragment col

  const __bf16* __restrict__ xrow = X + (size_t)row * K;
  const __bf16* __restrict__ wcol = W + (size_t)col * K;

  v8f acc = {};
  for (int k0 = 0; k0 < K; k0 += 32) {
    v8bf alo = *(const v8bf*)(xrow + k0 + 16 * grp);
    v8bf ahi = *(const v8bf*)(xrow + k0 + 16 * grp + 8);
    v16bf a;
#pragma unroll
    for (int j = 0; j < 8; ++j) { a[j] = alo[j]; a[j + 8] = ahi[j]; }
    v16bf b = *(const v16bf*)(wcol + k0 + 16 * grp);
    acc = __builtin_amdgcn_wmma_f32_16x16x32_bf16(
        /*neg_a=*/false, a, /*neg_b=*/false, b,
        /*c_mod=*/(short)0, acc, /*reuse_a=*/false, /*reuse_b=*/false);
  }

  const float bv = BIAS ? bias[col] : 0.f;
#pragma unroll
  for (int i = 0; i < 8; ++i) {
    int r = rt * 16 + i + 8 * grp;
    float v = acc[i] + bv;
    if (RELU) v = v > 0.f ? v : 0.f;
    if (Yf) Yf[(size_t)r * C + col] = v;
    if (Yb) Yb[(size_t)r * C + col] = f2bf(v);
  }
}

// ---------------------------------------------------------------------------
// Edge message + adjacency aggregation (fused), WMMA, bf16 operands.
// For node (b,v): msg[w,m] = sum_k x3[(b,v,w),k] * Cmat[(b,v),m*128+k]
// then Magg[b,w,m] += g[b,v,w] * (msg[w,m] + D[(b,v),m])
// One wave per (b,v, wtile(2), mtile(4)) -> 2048 blocks, K=128.
// ---------------------------------------------------------------------------
__global__ __launch_bounds__(32)
void edge_msg_kernel(const __bf16* __restrict__ x3, const __bf16* __restrict__ Cmat,
                     const float* __restrict__ D,  const float* __restrict__ g,
                     float* __restrict__ Magg)
{
  const int blk = blockIdx.x;            // 0..2047
  const int mt  = blk & 3;
  const int wt  = (blk >> 2) & 1;
  const int bv  = blk >> 3;              // 0..255  (b*32+v)
  const int b   = bv >> 5;

  const int lane = threadIdx.x;
  const int l15  = lane & 15;
  const int grp  = lane >> 4;
  const int row  = wt * 16 + l15;        // w for A fragment
  const int colm = mt * 16 + l15;        // m for B fragment

  const __bf16* __restrict__ xrow = x3   + ((size_t)bv * 32 + row) * 128;
  const __bf16* __restrict__ wcol = Cmat + (size_t)bv * 8192 + (size_t)colm * 128;

  v8f acc = {};
#pragma unroll
  for (int k0 = 0; k0 < 128; k0 += 32) {
    v8bf alo = *(const v8bf*)(xrow + k0 + 16 * grp);
    v8bf ahi = *(const v8bf*)(xrow + k0 + 16 * grp + 8);
    v16bf a;
#pragma unroll
    for (int j = 0; j < 8; ++j) { a[j] = alo[j]; a[j + 8] = ahi[j]; }
    v16bf bfr = *(const v16bf*)(wcol + k0 + 16 * grp);
    acc = __builtin_amdgcn_wmma_f32_16x16x32_bf16(false, a, false, bfr,
                                                  (short)0, acc, false, false);
  }

  const int m   = mt * 16 + l15;
  const float d = D[(size_t)bv * 64 + m];
#pragma unroll
  for (int i = 0; i < 8; ++i) {
    int w = wt * 16 + i + 8 * grp;
    float gv = g[(size_t)bv * 32 + w];                       // g[b,v,w]
    atomicAdd(&Magg[((size_t)(b * 32 + w)) * 64 + m], (acc[i] + d) * gv);
  }
}

// ---------------------------------------------------------------------------
// GRU step: h' = (1-z)*n + z*h ; gate order (r,z,n) ; masked.
// One block per node, 64 threads; maintains f32 state + bf16 mirror.
// ---------------------------------------------------------------------------
__global__ __launch_bounds__(64)
void gru_kernel(const float* __restrict__ Magg, float* __restrict__ hstate,
                __bf16* __restrict__ hstb,
                const float* __restrict__ Wih,  const float* __restrict__ Whh,
                const float* __restrict__ bih,  const float* __restrict__ bhh,
                const float* __restrict__ mask)
{
  const int node = blockIdx.x;     // 0..255
  const int t    = threadIdx.x;    // 0..63
  __shared__ float sm[64], sh[64];
  sm[t] = Magg[(size_t)node * 64 + t];
  sh[t] = hstate[(size_t)node * 64 + t];
  __syncthreads();

  float gi[3], gh[3];
#pragma unroll
  for (int gk = 0; gk < 3; ++gk) {
    int j = gk * 64 + t;
    float a = bih[j], c = bhh[j];
    for (int k = 0; k < 64; ++k) {
      a += Wih[(size_t)j * 64 + k] * sm[k];
      c += Whh[(size_t)j * 64 + k] * sh[k];
    }
    gi[gk] = a; gh[gk] = c;
  }
  float r = 1.f / (1.f + __expf(-(gi[0] + gh[0])));
  float z = 1.f / (1.f + __expf(-(gi[1] + gh[1])));
  float n = tanhf(gi[2] + r * gh[2]);
  float hnew = ((1.f - z) * n + z * sh[t]) * mask[node];
  hstate[(size_t)node * 64 + t] = hnew;
  hstb[(size_t)node * 64 + t]   = f2bf(hnew);
}

// ---------------------------------------------------------------------------
// Small helper kernels
// ---------------------------------------------------------------------------
__global__ void init_kernel(const float* __restrict__ h_in, float* __restrict__ h0,
                            float* __restrict__ hstate, __bf16* __restrict__ hstb,
                            float* __restrict__ mask)
{
  int node = blockIdx.x * blockDim.x + threadIdx.x;
  if (node >= 256) return;
  float s = 0.f;
  for (int f = 0; f < 32; ++f) s += h_in[(size_t)node * 32 + f];
  mask[node] = (s > 0.f) ? 1.f : 0.f;
  for (int h = 0; h < 64; ++h) {
    float v = (h < 32) ? h_in[(size_t)node * 32 + h] : 0.f;
    h0[(size_t)node * 64 + h]     = v;
    hstate[(size_t)node * 64 + h] = v;
    hstb[(size_t)node * 64 + h]   = f2bf(v);
  }
}

// W4:[4096,128] (row o=m*64+h, col k) -> W4r bf16:[8192,64] (row m*128+k, col h)
__global__ void transpose_w4(const float* __restrict__ W4, __bf16* __restrict__ W4r)
{
  int idx = blockIdx.x * blockDim.x + threadIdx.x;
  if (idx >= 4096 * 128) return;
  int o4 = idx >> 7;        // m*64+h
  int k  = idx & 127;
  int m  = o4 >> 6, h = o4 & 63;
  W4r[((size_t)((m << 7) + k)) * 64 + h] = f2bf(W4[idx]);
}

__global__ void cvt_bf16_kernel(const float* __restrict__ src, __bf16* __restrict__ dst, int n)
{
  int i = blockIdx.x * blockDim.x + threadIdx.x;
  if (i < n) dst[i] = f2bf(src[i]);
}

// src:[rs,cs] f32 -> dst:[rd,cd] bf16 zero-padded
__global__ void pad_cvt_kernel(const float* __restrict__ src, __bf16* __restrict__ dst,
                               int rs, int cs, int rd, int cd)
{
  int i = blockIdx.x * blockDim.x + threadIdx.x;
  if (i >= rd * cd) return;
  int r = i / cd, c = i % cd;
  dst[i] = (r < rs && c < cs) ? f2bf(src[(size_t)r * cs + c]) : f2bf(0.f);
}

__global__ void pad_f32_kernel(const float* __restrict__ src, float* __restrict__ dst,
                               int ns, int nd)
{
  int i = blockIdx.x * blockDim.x + threadIdx.x;
  if (i < nd) dst[i] = (i < ns) ? src[i] : 0.f;
}

__global__ void zero_kernel(float* __restrict__ p, int n)
{
  int i = blockIdx.x * blockDim.x + threadIdx.x;
  if (i < n) p[i] = 0.f;
}

// catb[n, 0:64] = h0 ; catb[n, 64:128] = hT   (bf16 for the i-MLP)
__global__ void cat_kernel(const float* __restrict__ h0, const float* __restrict__ h,
                           __bf16* __restrict__ catb)
{
  int i = blockIdx.x * blockDim.x + threadIdx.x;
  if (i >= 256 * 128) return;
  int n = i >> 7, c = i & 127;
  float v = (c < 64) ? h0[(size_t)n * 64 + c] : h[(size_t)n * 64 + (c - 64)];
  catb[i] = f2bf(v);
}

// out[b,t] = sum_v mask * sigmoid(gate_logit) * val   (gate/val padded to 16 cols)
__global__ void readout_kernel(const float* __restrict__ gate, const float* __restrict__ val,
                               const float* __restrict__ mask, float* __restrict__ out)
{
  int idx = blockIdx.x * blockDim.x + threadIdx.x;
  if (idx >= 96) return;
  int b = idx / 12, t = idx % 12;
  float s = 0.f;
  for (int v = 0; v < 32; ++v) {
    int n = b * 32 + v;
    float gl = gate[(size_t)n * 16 + t];
    float sg = 1.f / (1.f + __expf(-gl));
    s += mask[n] * sg * val[(size_t)n * 16 + t];
  }
  out[idx] = s;
}

// ---------------------------------------------------------------------------
// Host-side launcher
// ---------------------------------------------------------------------------
static inline void run_gemm(const __bf16* X, const __bf16* W, const float* bias,
                            float* Yf, __bf16* Yb, int R, int K, int C, bool relu,
                            hipStream_t s)
{
  dim3 grid((unsigned)(C / 16), (unsigned)(R / 16));
  if (relu)       gemm_bf16_wmma<true,  true ><<<grid, 32, 0, s>>>(X, W, bias, Yf, Yb, R, K, C);
  else if (bias)  gemm_bf16_wmma<false, true ><<<grid, 32, 0, s>>>(X, W, bias, Yf, Yb, R, K, C);
  else            gemm_bf16_wmma<false, false><<<grid, 32, 0, s>>>(X, W, nullptr, Yf, Yb, R, K, C);
}

static inline void run_cvt(const float* s, __bf16* d, int n, hipStream_t st)
{ cvt_bf16_kernel<<<(n + 255) / 256, 256, 0, st>>>(s, d, n); }

extern "C" void kernel_launch(void* const* d_in, const int* in_sizes, int n_in,
                              void* d_out, int out_size, void* d_ws, size_t ws_size,
                              hipStream_t stream)
{
  (void)in_sizes; (void)n_in; (void)out_size; (void)ws_size;

  const float* g    = (const float*)d_in[0];   // [8,32,32]
  const float* h_in = (const float*)d_in[1];   // [8,32,32]
  const float* e    = (const float*)d_in[2];   // [8,32,32,16]
  const float* mW1 = (const float*)d_in[3];  const float* mb1 = (const float*)d_in[4];
  const float* mW2 = (const float*)d_in[5];  const float* mb2 = (const float*)d_in[6];
  const float* mW3 = (const float*)d_in[7];  const float* mb3 = (const float*)d_in[8];
  const float* mW4 = (const float*)d_in[9];  const float* mb4 = (const float*)d_in[10];
  const float* Wih = (const float*)d_in[11]; const float* Whh = (const float*)d_in[12];
  const float* bih = (const float*)d_in[13]; const float* bhh = (const float*)d_in[14];
  const float* iW1 = (const float*)d_in[15]; const float* ib1 = (const float*)d_in[16];
  const float* iW2 = (const float*)d_in[17]; const float* ib2 = (const float*)d_in[18];
  const float* iW3 = (const float*)d_in[19]; const float* ib3 = (const float*)d_in[20];
  const float* iW4 = (const float*)d_in[21]; const float* ib4 = (const float*)d_in[22];
  const float* jW1 = (const float*)d_in[23]; const float* jb1 = (const float*)d_in[24];
  const float* jW2 = (const float*)d_in[25]; const float* jb2 = (const float*)d_in[26];
  const float* jW3 = (const float*)d_in[27]; const float* jb3 = (const float*)d_in[28];
  const float* jW4 = (const float*)d_in[29]; const float* jb4 = (const float*)d_in[30];

  // ---- workspace layout (bump allocator, 256B aligned) ----
  char* wp = (char*)d_ws;
  auto alloc = [&](size_t bytes) -> char* {
    char* r = wp; wp += (bytes + 255) & ~(size_t)255; return r;
  };
  __bf16* e_pad = (__bf16*)alloc(8192 * 32 * 2);        // e padded K=16->32
  __bf16* x1x3  = (__bf16*)alloc((size_t)8192 * 128 * 2); // x1, then x3
  __bf16* x2C   = (__bf16*)alloc((size_t)8192 * 256 * 2); // x2, then C [256,8192]
  __bf16* w1p   = (__bf16*)alloc(128 * 32 * 2);
  __bf16* w2b   = (__bf16*)alloc(256 * 128 * 2);
  __bf16* w3b   = (__bf16*)alloc(128 * 256 * 2);
  __bf16* w4rb  = (__bf16*)alloc((size_t)8192 * 64 * 2);
  __bf16* b4wb  = (__bf16*)alloc(4096 * 2);             // b4 as [64,64] weight
  __bf16* iw1b  = (__bf16*)alloc(128 * 128 * 2);
  __bf16* iw2b  = (__bf16*)alloc(256 * 128 * 2);
  __bf16* iw3b  = (__bf16*)alloc(128 * 256 * 2);
  __bf16* iw4p  = (__bf16*)alloc(16 * 128 * 2);         // [12,128]->[16,128]
  __bf16* jw1b  = (__bf16*)alloc(128 * 64 * 2);
  __bf16* jw2b  = (__bf16*)alloc(256 * 128 * 2);
  __bf16* jw3b  = (__bf16*)alloc(128 * 256 * 2);
  __bf16* jw4p  = (__bf16*)alloc(16 * 128 * 2);
  __bf16* hstb  = (__bf16*)alloc(256 * 64 * 2);
  __bf16* catb  = (__bf16*)alloc(256 * 128 * 2);
  __bf16* t0b   = (__bf16*)alloc(256 * 256 * 2);
  __bf16* t1b   = (__bf16*)alloc(256 * 256 * 2);
  float*  Dbuf  = (float*)alloc(256 * 64 * 4);
  float*  Magg  = (float*)alloc(256 * 64 * 4);
  float*  h0    = (float*)alloc(256 * 64 * 4);
  float*  hst   = (float*)alloc(256 * 64 * 4);
  float*  mask  = (float*)alloc(256 * 4);
  float*  gate  = (float*)alloc(256 * 16 * 4);
  float*  val   = (float*)alloc(256 * 16 * 4);
  float*  ib4p  = (float*)alloc(16 * 4);
  float*  jb4p  = (float*)alloc(16 * 4);
  // total ~9 MB

  // ---- one-time prep: masks, bf16 weight conversions, padding ----
  init_kernel<<<1, 256, 0, stream>>>(h_in, h0, hst, hstb, mask);
  transpose_w4<<<(4096 * 128 + 255) / 256, 256, 0, stream>>>(mW4, w4rb);
  pad_cvt_kernel<<<(8192 * 32 + 255) / 256, 256, 0, stream>>>(e, e_pad, 8192, 16, 8192, 32);
  pad_cvt_kernel<<<(128 * 32 + 255) / 256, 256, 0, stream>>>(mW1, w1p, 128, 16, 128, 32);
  run_cvt(mW2, w2b, 256 * 128, stream);
  run_cvt(mW3, w3b, 128 * 256, stream);
  run_cvt(mb4, b4wb, 4096, stream);
  run_cvt(iW1, iw1b, 128 * 128, stream);
  run_cvt(iW2, iw2b, 256 * 128, stream);
  run_cvt(iW3, iw3b, 128 * 256, stream);
  pad_cvt_kernel<<<(16 * 128 + 255) / 256, 256, 0, stream>>>(iW4, iw4p, 12, 128, 16, 128);
  run_cvt(jW1, jw1b, 128 * 64, stream);
  run_cvt(jW2, jw2b, 256 * 128, stream);
  run_cvt(jW3, jw3b, 128 * 256, stream);
  pad_cvt_kernel<<<(16 * 128 + 255) / 256, 256, 0, stream>>>(jW4, jw4p, 12, 128, 16, 128);
  pad_f32_kernel<<<1, 16, 0, stream>>>(ib4, ib4p, 12, 16);
  pad_f32_kernel<<<1, 16, 0, stream>>>(jb4, jb4p, 12, 16);

  // ---- edge MLP through layer 3 (x3 is all we ever need of the edge net) ----
  run_gemm(e_pad, w1p, mb1, nullptr, x1x3, 8192, 32,  128, true, stream);  // x1
  run_gemm(x1x3,  w2b, mb2, nullptr, x2C,  8192, 128, 256, true, stream);  // x2
  run_gemm(x2C,   w3b, mb3, nullptr, x1x3, 8192, 256, 128, true, stream);  // x3

  // ---- message-passing layers ----
  for (int layer = 0; layer < 3; ++layer) {
    // C[node, m*128+k] = sum_h hst[node,h] * W4[m*64+h,k]   (bf16 out)
    run_gemm(hstb, w4rb, nullptr, nullptr, x2C, 256, 64, 8192, false, stream);
    // D[node, m] = sum_h b4[m*64+h] * hst[node,h]           (f32 out)
    run_gemm(hstb, b4wb, nullptr, Dbuf, nullptr, 256, 64, 64, false, stream);
    zero_kernel<<<(16384 + 255) / 256, 256, 0, stream>>>(Magg, 16384);
    edge_msg_kernel<<<2048, 32, 0, stream>>>(x1x3, x2C, Dbuf, g, Magg);
    gru_kernel<<<256, 64, 0, stream>>>(Magg, hst, hstb, Wih, Whh, bih, bhh, mask);
  }

  // ---- readout ----
  cat_kernel<<<(256 * 128 + 255) / 256, 256, 0, stream>>>(h0, hst, catb);
  // gate net i: [256,128] -> 128 -> 256 -> 128 -> 16(pad of 12) logits
  run_gemm(catb, iw1b, ib1,  nullptr, t0b,  256, 128, 128, true,  stream);
  run_gemm(t0b,  iw2b, ib2,  nullptr, t1b,  256, 128, 256, true,  stream);
  run_gemm(t1b,  iw3b, ib3,  nullptr, t0b,  256, 256, 128, true,  stream);
  run_gemm(t0b,  iw4p, ib4p, gate,    nullptr, 256, 128, 16, false, stream);
  // value net j: [256,64] -> 128 -> 256 -> 128 -> 16(pad of 12)
  run_gemm(hstb, jw1b, jb1,  nullptr, t0b,  256, 64,  128, true,  stream);
  run_gemm(t0b,  jw2b, jb2,  nullptr, t1b,  256, 128, 256, true,  stream);
  run_gemm(t1b,  jw3b, jb3,  nullptr, t0b,  256, 256, 128, true,  stream);
  run_gemm(t0b,  jw4p, jb4p, val,     nullptr, 256, 128, 16, false, stream);

  readout_kernel<<<1, 96, 0, stream>>>(gate, val, mask, (float*)d_out);
}
